// Transformer_39926015984080
// MI455X (gfx1250) — compile-verified, matrix operand order checked by reading, not run on hardware
//
#include <hip/hip_runtime.h>
#include <hip/hip_bf16.h>

// ---------------- model constants ----------------
#define BB   2
#define TT   1024
#define DD   768
#define FF   3072
#define NH   12
#define HDIM 64
#define VV   50257
#define NL   4
#define BT   (BB*TT)      // 2048
#define QKV  (3*DD)       // 2304
#define BH   (BB*NH)      // 24

typedef __bf16 bf16_t;
typedef __attribute__((ext_vector_type(16))) __bf16 v16bf;
typedef __attribute__((ext_vector_type(8)))  __bf16 v8bf;
typedef __attribute__((ext_vector_type(8)))  float  v8f;

// ---------------- WMMA helpers ----------------
__device__ __forceinline__ v8f zero_v8f() {
    v8f z = {0.f,0.f,0.f,0.f,0.f,0.f,0.f,0.f};
    return z;
}

// Load a 16x32 bf16 A-fragment (or column-major B-fragment) for this lane.
// p must already point at element [row_or_col, k + half*8]; +16 elements gives
// the second contiguous 16-byte chunk (K = k+16+half*8 .. ).
__device__ __forceinline__ v16bf load_frag(const bf16_t* __restrict__ p) {
    v8bf lo = *reinterpret_cast<const v8bf*>(p);
    v8bf hi = *reinterpret_cast<const v8bf*>(p + 16);
    v16bf f;
#pragma unroll
    for (int i = 0; i < 8; ++i) { f[i] = lo[i]; f[i + 8] = hi[i]; }
    return f;
}

__device__ __forceinline__ v8f wmma_bf16(v16bf a, v16bf b, v8f c) {
    // (neg_a, A, neg_b, B, c_mod, C, reuse_a, reuse_b)
    return __builtin_amdgcn_wmma_f32_16x16x32_bf16(false, a, false, b,
                                                   (short)0, c, false, false);
}

__device__ __forceinline__ float gelu_f(float x) {
    const float c = 0.7978845608028654f;   // sqrt(2/pi)
    float x3 = x * x * x;
    return 0.5f * x * (1.0f + tanhf(c * (x + 0.044715f * x3)));
}

// ---------------- weight conversion ----------------
// W [K,N] fp32 row-major  ->  Wt [N,K] bf16 row-major (K,N multiples of 32)
__global__ __launch_bounds__(256)
void convert_transpose_kernel(const float* __restrict__ W, bf16_t* __restrict__ Wt,
                              int K, int N) {
    __shared__ float tile[32][33];
    int n0 = blockIdx.x * 32, k0 = blockIdx.y * 32;
    int tx = threadIdx.x, ty = threadIdx.y;    // 32 x 8
#pragma unroll
    for (int j = 0; j < 4; ++j)
        tile[ty + j * 8][tx] = W[(size_t)(k0 + ty + j * 8) * N + n0 + tx];
    __syncthreads();
#pragma unroll
    for (int j = 0; j < 4; ++j)
        Wt[(size_t)(n0 + ty + j * 8) * K + k0 + tx] = (bf16_t)tile[tx][ty + j * 8];
}

// head_w is already [V, D] = [N, K]: straight fp32 -> bf16 copy
__global__ void convert_bf16_kernel(const float* __restrict__ src,
                                    bf16_t* __restrict__ dst, size_t n) {
    size_t i = (size_t)blockIdx.x * blockDim.x + threadIdx.x;
    if (i < n) dst[i] = (bf16_t)src[i];
}

// ---------------- embedding ----------------
__global__ void embed_kernel(const int* __restrict__ tokens,
                             const float* __restrict__ wte,
                             const float* __restrict__ wpe,
                             float* __restrict__ x) {
    size_t i = (size_t)blockIdx.x * blockDim.x + threadIdx.x;
    if (i >= (size_t)BT * DD) return;
    int m = (int)(i / DD), d = (int)(i % DD);
    int t = m & (TT - 1);
    int tok = tokens[m];
    x[i] = wte[(size_t)tok * DD + d] + wpe[(size_t)t * DD + d];
}

// ---------------- layernorm (fp32 in -> bf16 out) ----------------
__global__ __launch_bounds__(256)
void layernorm_kernel(const float* __restrict__ x, const float* __restrict__ w,
                      const float* __restrict__ b, bf16_t* __restrict__ out) {
    int m = blockIdx.x;
    int tid = threadIdx.x;
    const float* xr = x + (size_t)m * DD;
    __shared__ float r1[256], r2[256];
    float s = 0.f, s2 = 0.f;
    for (int i = tid; i < DD; i += 256) { float v = xr[i]; s += v; s2 += v * v; }
    r1[tid] = s; r2[tid] = s2;
    __syncthreads();
    for (int st = 128; st > 0; st >>= 1) {
        if (tid < st) { r1[tid] += r1[tid + st]; r2[tid] += r2[tid + st]; }
        __syncthreads();
    }
    float mean = r1[0] * (1.0f / DD);
    float var  = r2[0] * (1.0f / DD) - mean * mean;
    float rstd = rsqrtf(var + 1e-5f);
    bf16_t* orow = out + (size_t)m * DD;
    for (int i = tid; i < DD; i += 256)
        orow[i] = (bf16_t)((xr[i] - mean) * rstd * w[i] + b[i]);
}

// ---------------- generic bf16 WMMA GEMM ----------------
// C[M,N] = A[M,K](bf16, row-major, lda) x Bw[N,K](bf16, row-major) + epilogue
// Block: 256 threads = 8 waves, arranged 4(m) x 2(n); wave tile 32x64.
// Block tile: 128(M) x 128(N).  Per K=32 step: 2 A frags + 4 B frags, 8 WMMAs.
// EPI 0: out bf16 = acc + bias                  (qkv)
// EPI 1: out f32 += acc + bias (residual)       (proj, fc2)
// EPI 2: out bf16 = gelu(acc + bias)            (fc)
// EPI 3: out f32  = acc, N-bounds masked        (lm head)
template <int EPI>
__global__ __launch_bounds__(256)
void gemm_bf16_kernel(const bf16_t* __restrict__ A, int lda,
                      const bf16_t* __restrict__ Bw,
                      const float* __restrict__ bias,
                      float* __restrict__ outF, bf16_t* __restrict__ outB,
                      int N, int K) {
    int tid  = threadIdx.x;
    int lane = tid & 31, wid = tid >> 5;
    int l16  = lane & 15, half = lane >> 4;
    int wm   = wid >> 1, wn = wid & 1;       // 4 x 2 waves
    int row0 = blockIdx.y * 128 + wm * 32;
    int col0 = blockIdx.x * 128 + wn * 64;

    v8f acc[2][4];
#pragma unroll
    for (int i = 0; i < 2; ++i)
#pragma unroll
        for (int j = 0; j < 4; ++j) acc[i][j] = zero_v8f();

    const bf16_t* pa0 = A + (size_t)(row0 + l16) * lda + half * 8;
    const bf16_t* pa1 = pa0 + (size_t)16 * lda;
    const bf16_t* pb[4];
#pragma unroll
    for (int j = 0; j < 4; ++j) {
        int c = col0 + j * 16 + l16;
        if (c >= N) c = N - 1;               // clamp (only matters for head)
        pb[j] = Bw + (size_t)c * K + half * 8;
    }

    for (int k = 0; k < K; k += 32) {
        __builtin_prefetch(pb[0] + k + 256, 0, 1);
        __builtin_prefetch(pb[2] + k + 256, 0, 1);
        v16bf a0 = load_frag(pa0 + k);
        v16bf a1 = load_frag(pa1 + k);
        v16bf b0 = load_frag(pb[0] + k);
        v16bf b1 = load_frag(pb[1] + k);
        v16bf b2 = load_frag(pb[2] + k);
        v16bf b3 = load_frag(pb[3] + k);
        acc[0][0] = wmma_bf16(a0, b0, acc[0][0]);
        acc[1][0] = wmma_bf16(a1, b0, acc[1][0]);
        acc[0][1] = wmma_bf16(a0, b1, acc[0][1]);
        acc[1][1] = wmma_bf16(a1, b1, acc[1][1]);
        acc[0][2] = wmma_bf16(a0, b2, acc[0][2]);
        acc[1][2] = wmma_bf16(a1, b2, acc[1][2]);
        acc[0][3] = wmma_bf16(a0, b3, acc[0][3]);
        acc[1][3] = wmma_bf16(a1, b3, acc[1][3]);
    }

#pragma unroll
    for (int i = 0; i < 2; ++i) {
#pragma unroll
        for (int j = 0; j < 4; ++j) {
            int colw = col0 + j * 16 + l16;
            float bv = 0.f;
            if (EPI != 3) bv = bias[colw];
#pragma unroll
            for (int r = 0; r < 8; ++r) {
                int rw = row0 + i * 16 + half * 8 + r;
                size_t idx = (size_t)rw * N + colw;
                float v = acc[i][j][r] + bv;
                if (EPI == 0) outB[idx] = (bf16_t)v;
                else if (EPI == 1) outF[idx] = outF[idx] + v;
                else if (EPI == 2) outB[idx] = (bf16_t)gelu_f(v);
                else { if (colw < N) outF[idx] = v; }
            }
        }
    }
}

// ---------------- attention: v transpose [b,h,d,t] ----------------
__global__ void v_transpose_kernel(const bf16_t* __restrict__ qkv,
                                   bf16_t* __restrict__ vt) {
    size_t i = (size_t)blockIdx.x * blockDim.x + threadIdx.x;
    if (i >= (size_t)BH * HDIM * TT) return;
    int t = (int)(i & (TT - 1));
    size_t r = i >> 10;
    int d = (int)(r & (HDIM - 1)); r >>= 6;
    int h = (int)(r % NH);
    int b = (int)(r / NH);
    vt[i] = qkv[(size_t)(b * TT + t) * QKV + 2 * DD + h * HDIM + d];
}

// ---------------- attention: scores = q @ k^T * 0.125 (causal-tiled) -------
__global__ __launch_bounds__(256)
void attn_scores_kernel(const bf16_t* __restrict__ qkv,
                        float* __restrict__ scores) {
    int bh = blockIdx.z;
    int b = bh / NH, h = bh % NH;
    int q0 = blockIdx.y * 64;
    int k0 = blockIdx.x * 128;
    if (k0 > q0 + 63) return;   // fully masked tile

    int tid  = threadIdx.x;
    int lane = tid & 31, wid = tid >> 5;
    int l16  = lane & 15, half = lane >> 4;
    int wm   = wid >> 2, wn = wid & 3;
    int row0 = q0 + wm * 32;          // query rows
    int col0 = k0 + wn * 32;          // key cols

    v8f acc[2][2];
#pragma unroll
    for (int i = 0; i < 2; ++i)
#pragma unroll
        for (int j = 0; j < 2; ++j) acc[i][j] = zero_v8f();

    const bf16_t* pa0 = qkv + (size_t)(b * TT + row0 + l16) * QKV + h * HDIM + half * 8;
    const bf16_t* pa1 = pa0 + (size_t)16 * QKV;
    const bf16_t* pb0 = qkv + (size_t)(b * TT + col0 + l16) * QKV + DD + h * HDIM + half * 8;
    const bf16_t* pb1 = pb0 + (size_t)16 * QKV;

#pragma unroll
    for (int k = 0; k < HDIM; k += 32) {
        v16bf a0 = load_frag(pa0 + k);
        v16bf a1 = load_frag(pa1 + k);
        v16bf b0 = load_frag(pb0 + k);
        v16bf b1 = load_frag(pb1 + k);
        acc[0][0] = wmma_bf16(a0, b0, acc[0][0]);
        acc[0][1] = wmma_bf16(a0, b1, acc[0][1]);
        acc[1][0] = wmma_bf16(a1, b0, acc[1][0]);
        acc[1][1] = wmma_bf16(a1, b1, acc[1][1]);
    }

#pragma unroll
    for (int i = 0; i < 2; ++i)
#pragma unroll
        for (int j = 0; j < 2; ++j) {
            int colw = col0 + j * 16 + l16;
#pragma unroll
            for (int r = 0; r < 8; ++r) {
                int rw = row0 + i * 16 + half * 8 + r;
                scores[((size_t)bh * TT + rw) * TT + colw] = acc[i][j][r] * 0.125f;
            }
        }
}

// ---------------- attention: causal softmax row, bf16 probs ----------------
__global__ __launch_bounds__(256)
void softmax_kernel(const float* __restrict__ scores, bf16_t* __restrict__ probs) {
    int r = blockIdx.x;                  // r = bh*T + q
    int q = r & (TT - 1);
    int tid = threadIdx.x;
    const float* s = scores + (size_t)r * TT;
    bf16_t* p = probs + (size_t)r * TT;
    int len = q + 1;

    __shared__ float red[256];
    float mx = -3.0e38f;
    for (int i = tid; i < len; i += 256) mx = fmaxf(mx, s[i]);
    red[tid] = mx; __syncthreads();
    for (int st = 128; st > 0; st >>= 1) {
        if (tid < st) red[tid] = fmaxf(red[tid], red[tid + st]);
        __syncthreads();
    }
    mx = red[0]; __syncthreads();

    float sum = 0.f;
    for (int i = tid; i < len; i += 256) sum += expf(s[i] - mx);
    red[tid] = sum; __syncthreads();
    for (int st = 128; st > 0; st >>= 1) {
        if (tid < st) red[tid] += red[tid + st];
        __syncthreads();
    }
    float inv = 1.0f / red[0];

    for (int i = tid; i < len; i += 256) p[i] = (bf16_t)(expf(s[i] - mx) * inv);
    for (int i = len + tid; i < TT; i += 256) p[i] = (bf16_t)0.0f;   // mask tail
}

// ---------------- attention: o = probs @ v  (vt is [bh, d, t]) -------------
__global__ __launch_bounds__(128)
void attn_v_kernel(const bf16_t* __restrict__ probs, const bf16_t* __restrict__ vt,
                   bf16_t* __restrict__ o) {
    int bh = blockIdx.y;
    int b = bh / NH, h = bh % NH;
    int q0 = blockIdx.x * 64;

    int tid  = threadIdx.x;
    int lane = tid & 31, wid = tid >> 5;
    int l16  = lane & 15, half = lane >> 4;
    int wm   = wid >> 1, wn = wid & 1;
    int row0 = q0 + wm * 32;
    int col0 = wn * 32;

    v8f acc[2][2];
#pragma unroll
    for (int i = 0; i < 2; ++i)
#pragma unroll
        for (int j = 0; j < 2; ++j) acc[i][j] = zero_v8f();

    const bf16_t* pa0 = probs + ((size_t)bh * TT + row0 + l16) * TT + half * 8;
    const bf16_t* pa1 = pa0 + (size_t)16 * TT;
    const bf16_t* pb0 = vt + (size_t)(bh * HDIM + col0 + l16) * TT + half * 8;
    const bf16_t* pb1 = pb0 + (size_t)16 * TT;

    int kmax = q0 + 64;   // probs are zero beyond the causal boundary
    for (int k = 0; k < kmax; k += 32) {
        v16bf a0 = load_frag(pa0 + k);
        v16bf a1 = load_frag(pa1 + k);
        v16bf b0 = load_frag(pb0 + k);
        v16bf b1 = load_frag(pb1 + k);
        acc[0][0] = wmma_bf16(a0, b0, acc[0][0]);
        acc[0][1] = wmma_bf16(a0, b1, acc[0][1]);
        acc[1][0] = wmma_bf16(a1, b0, acc[1][0]);
        acc[1][1] = wmma_bf16(a1, b1, acc[1][1]);
    }

#pragma unroll
    for (int i = 0; i < 2; ++i)
#pragma unroll
        for (int j = 0; j < 2; ++j) {
            int colw = col0 + j * 16 + l16;      // head-dim index
#pragma unroll
            for (int r = 0; r < 8; ++r) {
                int rw = row0 + i * 16 + half * 8 + r;
                o[(size_t)(b * TT + rw) * DD + h * HDIM + colw] = (bf16_t)acc[i][j][r];
            }
        }
}

// ---------------- host launcher ----------------
extern "C" void kernel_launch(void* const* d_in, const int* in_sizes, int n_in,
                              void* d_out, int out_size, void* d_ws, size_t ws_size,
                              hipStream_t stream) {
    const int*   tokens = (const int*)  d_in[0];
    const float* wte    = (const float*)d_in[1];
    const float* wpe    = (const float*)d_in[2];
    const float* ln1_w  = (const float*)d_in[3];
    const float* ln1_b  = (const float*)d_in[4];
    const float* attn_w = (const float*)d_in[5];
    const float* attn_b = (const float*)d_in[6];
    const float* proj_w = (const float*)d_in[7];
    const float* proj_b = (const float*)d_in[8];
    const float* ln2_w  = (const float*)d_in[9];
    const float* ln2_b  = (const float*)d_in[10];
    const float* fc_w   = (const float*)d_in[11];
    const float* fc_b   = (const float*)d_in[12];
    const float* fc2_w  = (const float*)d_in[13];
    const float* fc2_b  = (const float*)d_in[14];
    const float* lnf_w  = (const float*)d_in[15];
    const float* lnf_b  = (const float*)d_in[16];
    const float* head_w = (const float*)d_in[17];

    char* ws = (char*)d_ws;
    size_t off = 0;
    auto alloc = [&](size_t bytes) {
        size_t o = off;
        off += (bytes + 255) & ~(size_t)255;
        return o;
    };
    bf16_t* wt_attn = (bf16_t*)(ws + alloc((size_t)NL * QKV * DD * 2));
    bf16_t* wt_proj = (bf16_t*)(ws + alloc((size_t)NL * DD * DD * 2));
    bf16_t* wt_fc   = (bf16_t*)(ws + alloc((size_t)NL * FF * DD * 2));
    bf16_t* wt_fc2  = (bf16_t*)(ws + alloc((size_t)NL * DD * FF * 2));
    bf16_t* head_b  = (bf16_t*)(ws + alloc((size_t)VV * DD * 2));
    float*  xbuf    = (float*) (ws + alloc((size_t)BT * DD * 4));
    bf16_t* hbuf    = (bf16_t*)(ws + alloc((size_t)BT * DD * 2));
    bf16_t* qkvb    = (bf16_t*)(ws + alloc((size_t)BT * QKV * 2));
    bf16_t* vtb     = (bf16_t*)(ws + alloc((size_t)BH * HDIM * TT * 2));
    float*  scb     = (float*) (ws + alloc((size_t)BH * TT * TT * 4));
    bf16_t* prb     = (bf16_t*)(ws + alloc((size_t)BH * TT * TT * 2));
    bf16_t* obuf    = (bf16_t*)(ws + alloc((size_t)BT * DD * 2));
    bf16_t* gbuf    = (bf16_t*)(ws + alloc((size_t)BT * FF * 2));

    dim3 tb32(32, 8);
    // ---- per-launch weight conversion (fp32 [K,N] -> bf16 [N,K]) ----
    for (int l = 0; l < NL; ++l) {
        convert_transpose_kernel<<<dim3(QKV / 32, DD / 32), tb32, 0, stream>>>(
            attn_w + (size_t)l * DD * QKV, wt_attn + (size_t)l * QKV * DD, DD, QKV);
        convert_transpose_kernel<<<dim3(DD / 32, DD / 32), tb32, 0, stream>>>(
            proj_w + (size_t)l * DD * DD, wt_proj + (size_t)l * DD * DD, DD, DD);
        convert_transpose_kernel<<<dim3(FF / 32, DD / 32), tb32, 0, stream>>>(
            fc_w + (size_t)l * DD * FF, wt_fc + (size_t)l * FF * DD, DD, FF);
        convert_transpose_kernel<<<dim3(DD / 32, FF / 32), tb32, 0, stream>>>(
            fc2_w + (size_t)l * FF * DD, wt_fc2 + (size_t)l * DD * FF, FF, DD);
    }
    {
        size_t n = (size_t)VV * DD;
        convert_bf16_kernel<<<(unsigned)((n + 255) / 256), 256, 0, stream>>>(head_w, head_b, n);
    }

    // ---- embedding ----
    {
        size_t n = (size_t)BT * DD;
        embed_kernel<<<(unsigned)((n + 255) / 256), 256, 0, stream>>>(tokens, wte, wpe, xbuf);
    }

    // ---- transformer blocks ----
    for (int l = 0; l < NL; ++l) {
        const bf16_t* wa = wt_attn + (size_t)l * QKV * DD;
        const bf16_t* wp = wt_proj + (size_t)l * DD * DD;
        const bf16_t* wf = wt_fc   + (size_t)l * FF * DD;
        const bf16_t* w2 = wt_fc2  + (size_t)l * DD * FF;

        layernorm_kernel<<<BT, 256, 0, stream>>>(xbuf, ln1_w + l * DD, ln1_b + l * DD, hbuf);

        gemm_bf16_kernel<0><<<dim3(QKV / 128, BT / 128), 256, 0, stream>>>(
            hbuf, DD, wa, attn_b + (size_t)l * QKV, nullptr, qkvb, QKV, DD);

        {
            size_t n = (size_t)BH * HDIM * TT;
            v_transpose_kernel<<<(unsigned)((n + 255) / 256), 256, 0, stream>>>(qkvb, vtb);
        }

        attn_scores_kernel<<<dim3(TT / 128, TT / 64, BH), 256, 0, stream>>>(qkvb, scb);
        softmax_kernel<<<BH * TT, 256, 0, stream>>>(scb, prb);
        attn_v_kernel<<<dim3(TT / 64, BH), 128, 0, stream>>>(prb, vtb, obuf);

        gemm_bf16_kernel<1><<<dim3(DD / 128, BT / 128), 256, 0, stream>>>(
            obuf, DD, wp, proj_b + (size_t)l * DD, xbuf, nullptr, DD, DD);

        layernorm_kernel<<<BT, 256, 0, stream>>>(xbuf, ln2_w + l * DD, ln2_b + l * DD, hbuf);

        gemm_bf16_kernel<2><<<dim3(FF / 128, BT / 128), 256, 0, stream>>>(
            hbuf, DD, wf, fc_b + (size_t)l * FF, nullptr, gbuf, FF, DD);

        gemm_bf16_kernel<1><<<dim3(DD / 128, BT / 128), 256, 0, stream>>>(
            gbuf, FF, w2, fc2_b + (size_t)l * DD, xbuf, nullptr, DD, FF);
    }

    // ---- final LN + LM head ----
    layernorm_kernel<<<BT, 256, 0, stream>>>(xbuf, lnf_w, lnf_b, hbuf);
    gemm_bf16_kernel<3><<<dim3((VV + 127) / 128, BT / 128), 256, 0, stream>>>(
        hbuf, DD, head_b, nullptr, (float*)d_out, nullptr, VV, DD);

    (void)in_sizes; (void)n_in; (void)out_size; (void)ws_size;
}
